// GraphSAGE_32942399160960
// MI455X (gfx1250) — matrix operand order repaired and checked by reading, not compiled, not run on hardware
//
#include <hip/hip_runtime.h>
#include <hip/hip_bf16.h>
#include <stdint.h>
#include <stddef.h>

#define N_NODES   100000
#define N_EDGES   3200000
#define D_FEAT    128
#define D_HID     256
#define N_CLASSES 64

typedef __attribute__((ext_vector_type(16))) __bf16 bf16x16;
typedef __attribute__((ext_vector_type(8)))  __bf16 bf16x8;
typedef __attribute__((ext_vector_type(8)))  float  f32x8;

// ---------------------------------------------------------------- utilities

__global__ void zero_f32_kernel(float* __restrict__ p, size_t n) {
  size_t i = (size_t)blockIdx.x * blockDim.x + threadIdx.x;
  size_t stride = (size_t)gridDim.x * blockDim.x;
  for (; i < n; i += stride) p[i] = 0.0f;
}

__global__ void cvt_bf16_kernel(const float* __restrict__ in,
                                __bf16* __restrict__ out, size_t n) {
  size_t i = (size_t)blockIdx.x * blockDim.x + threadIdx.x;
  size_t stride = (size_t)gridDim.x * blockDim.x;
  for (; i < n; i += stride) out[i] = (__bf16)in[i];
}

// ---------------------------------------------------------------- degrees

__global__ void deg_kernel(const int* __restrict__ dst, float* __restrict__ deg, int ne) {
  int e = blockIdx.x * blockDim.x + threadIdx.x;
  if (e < ne) atomicAdd(&deg[dst[e]], 1.0f);
}

__global__ void deginv_kernel(float* __restrict__ deg, int n) {
  int i = blockIdx.x * blockDim.x + threadIdx.x;
  if (i < n) {
    float d = deg[i];
    deg[i] = (d > 0.0f) ? (1.0f / d) : 0.0f;
  }
}

// ------------------------------------------------- edge scatter (segment sum)
// One wave (32 lanes) per edge. Working set fits in 192MB L2, so gathers hit
// L2 and the f32 atomic-adds (no return -> STOREcnt path) ride L2 atomic units.

__global__ void scatter_f32_kernel(const int* __restrict__ src, const int* __restrict__ dst,
                                   const float* __restrict__ x, float* __restrict__ agg, int ne) {
  long long t = (long long)blockIdx.x * blockDim.x + threadIdx.x;
  int e = (int)(t >> 5);
  if (e >= ne) return;
  int lane = (int)(t & 31);            // 4 floats per lane, 128 per edge
  int s = src[e];
  int d = dst[e];
  float4 v = *(const float4*)(x + (size_t)s * D_FEAT + lane * 4);
  float* a = agg + (size_t)d * D_FEAT + lane * 4;
  atomicAdd(a + 0, v.x);
  atomicAdd(a + 1, v.y);
  atomicAdd(a + 2, v.z);
  atomicAdd(a + 3, v.w);
}

__global__ void scatter_bf16_kernel(const int* __restrict__ src, const int* __restrict__ dst,
                                    const __bf16* __restrict__ h, float* __restrict__ agg, int ne) {
  long long t = (long long)blockIdx.x * blockDim.x + threadIdx.x;
  int e = (int)(t >> 5);
  if (e >= ne) return;
  int lane = (int)(t & 31);            // 8 bf16 per lane, 256 per edge
  int s = src[e];
  int d = dst[e];
  bf16x8 v = *(const bf16x8*)(h + (size_t)s * D_HID + lane * 8);
  float* a = agg + (size_t)d * D_HID + lane * 8;
#pragma unroll
  for (int j = 0; j < 8; ++j) atomicAdd(a + j, (float)v[j]);
}

// ------------------------------------------- mean-normalize + convert to bf16

template <int D>
__global__ void scale_cvt_kernel(const float* __restrict__ agg, const float* __restrict__ dinv,
                                 __bf16* __restrict__ out, size_t n) {
  size_t i = (size_t)blockIdx.x * blockDim.x + threadIdx.x;
  size_t stride = (size_t)gridDim.x * blockDim.x;
  for (; i < n; i += stride) {
    size_t node = i / D;               // D is a power of two -> shift
    out[i] = (__bf16)(agg[i] * dinv[node]);
  }
}

// ---------------------------------------------------------------- WMMA GEMM
// out[N,M] = Aagg[N,K]·Wl[M,K]^T + Aroot[N,K]·Wr[M,K]^T + bias[M]  (+ReLU)
// One wave computes one 16x16 output tile with v_wmma_f32_16x16x32_bf16.
//
// Fragment layouts used here (wave32; A/B use contiguous k order, not the ISA table order):
//  A 16x32 bf16: lane row = lane&15; elems 0..15 -> K = (lane<16?0:16)+i
//  B 32x16 bf16: lane col = lane&15; elems 0..15 -> K = (lane<16?0:16)+i
//  C/D 16x16 f32: lane col = lane&15; VGPR j -> row = j + (lane<16?0:8)

template <int K, int M, bool RELU, bool OUT_BF16>
__global__ void sage_gemm_kernel(const __bf16* __restrict__ Aagg,
                                 const __bf16* __restrict__ Aroot,
                                 const __bf16* __restrict__ Wl,
                                 const __bf16* __restrict__ Wr,
                                 const float* __restrict__ bias,
                                 void* __restrict__ out) {
  const int lane = threadIdx.x & 31;
  const int wave = threadIdx.x >> 5;
  const int wavesPerBlock = blockDim.x >> 5;
  const int row0 = blockIdx.x * 16;                          // node tile
  const int col0 = (blockIdx.y * wavesPerBlock + wave) * 16; // output-col tile
  const int m  = lane & 15;
  const int hi = lane >> 4;

  f32x8 c = {0.0f, 0.0f, 0.0f, 0.0f, 0.0f, 0.0f, 0.0f, 0.0f};

  // ---- Aagg * Wl^T ----
  {
    const __bf16* arow = Aagg + (size_t)(row0 + m) * K + hi * 16;
    const __bf16* wrow = Wl   + (size_t)(col0 + m) * K + hi * 16;
#pragma unroll
    for (int kb = 0; kb < K; kb += 32) {
      bf16x8 alo = *(const bf16x8*)(arow + kb);
      bf16x8 ahi = *(const bf16x8*)(arow + kb + 8);
      bf16x16 a = __builtin_shufflevector(alo, ahi,
          0, 1, 2, 3, 4, 5, 6, 7, 8, 9, 10, 11, 12, 13, 14, 15);
      bf16x16 b = *(const bf16x16*)(wrow + kb);
      c = __builtin_amdgcn_wmma_f32_16x16x32_bf16(
              /*neg_a=*/false, a, /*neg_b=*/false, b,
              /*c_mod=*/(short)0, c, /*reuse_a=*/false, /*reuse_b=*/false);
    }
  }
  // ---- Aroot * Wr^T ----
  {
    const __bf16* arow = Aroot + (size_t)(row0 + m) * K + hi * 16;
    const __bf16* wrow = Wr    + (size_t)(col0 + m) * K + hi * 16;
#pragma unroll
    for (int kb = 0; kb < K; kb += 32) {
      bf16x8 alo = *(const bf16x8*)(arow + kb);
      bf16x8 ahi = *(const bf16x8*)(arow + kb + 8);
      bf16x16 a = __builtin_shufflevector(alo, ahi,
          0, 1, 2, 3, 4, 5, 6, 7, 8, 9, 10, 11, 12, 13, 14, 15);
      bf16x16 b = *(const bf16x16*)(wrow + kb);
      c = __builtin_amdgcn_wmma_f32_16x16x32_bf16(
              false, a, false, b, (short)0, c, false, false);
    }
  }

  const float bv = bias[col0 + m];
#pragma unroll
  for (int j = 0; j < 8; ++j) {
    float v = c[j] + bv;
    if (RELU) v = fmaxf(v, 0.0f);
    const size_t row = (size_t)(row0 + hi * 8 + j);
    if (OUT_BF16) ((__bf16*)out)[row * M + col0 + m] = (__bf16)v;
    else          ((float*)out)[row * M + col0 + m] = v;
  }
}

// ---------------------------------------------------------------- launcher

extern "C" void kernel_launch(void* const* d_in, const int* in_sizes, int n_in,
                              void* d_out, int out_size, void* d_ws, size_t ws_size,
                              hipStream_t stream) {
  const float* x     = (const float*)d_in[0];
  const int*   edges = (const int*)d_in[1];     // [2, N_EDGES]
  const float* W1_l  = (const float*)d_in[2];   // [256,128]
  const float* W1_r  = (const float*)d_in[3];
  const float* b1    = (const float*)d_in[4];
  const float* W2_l  = (const float*)d_in[5];   // [64,256]
  const float* W2_r  = (const float*)d_in[6];
  const float* b2    = (const float*)d_in[7];
  const int* src = edges;
  const int* dst = edges + N_EDGES;

  // ---- workspace carving (256B aligned slices) ----
  char* base = (char*)d_ws;
  size_t off = 0;
  auto take = [&](size_t bytes) -> char* {
    char* p = base + off;
    off = (off + bytes + 255) & ~(size_t)255;
    return p;
  };
  float*  deg    = (float*)take((size_t)N_NODES * 4);                 // deg -> deg_inv
  __bf16* x_bf   = (__bf16*)take((size_t)N_NODES * D_FEAT * 2);       // 25.6 MB
  __bf16* h_bf   = (__bf16*)take((size_t)N_NODES * D_HID * 2);        // 51.2 MB
  float*  aggf   = (float*)take((size_t)N_NODES * D_HID * 4);         // 102.4 MB (shared L1/L2 use)
  __bf16* agg_bf = (__bf16*)take((size_t)N_NODES * D_HID * 2);        // 51.2 MB
  __bf16* w1l_bf = (__bf16*)take((size_t)D_HID * D_FEAT * 2);
  __bf16* w1r_bf = (__bf16*)take((size_t)D_HID * D_FEAT * 2);
  __bf16* w2l_bf = (__bf16*)take((size_t)N_CLASSES * D_HID * 2);
  __bf16* w2r_bf = (__bf16*)take((size_t)N_CLASSES * D_HID * 2);
  (void)ws_size; (void)n_in; (void)in_sizes; (void)out_size;

  const int ZB = 4096;  // grid-stride blocks for streaming kernels

  // ---- prep: zero accumulators, bf16 conversions ----
  zero_f32_kernel<<<ZB, 256, 0, stream>>>(deg, (size_t)N_NODES);
  zero_f32_kernel<<<ZB, 256, 0, stream>>>(aggf, (size_t)N_NODES * D_FEAT);
  cvt_bf16_kernel<<<ZB, 256, 0, stream>>>(x, x_bf, (size_t)N_NODES * D_FEAT);
  cvt_bf16_kernel<<<128, 256, 0, stream>>>(W1_l, w1l_bf, (size_t)D_HID * D_FEAT);
  cvt_bf16_kernel<<<128, 256, 0, stream>>>(W1_r, w1r_bf, (size_t)D_HID * D_FEAT);
  cvt_bf16_kernel<<<64, 256, 0, stream>>>(W2_l, w2l_bf, (size_t)N_CLASSES * D_HID);
  cvt_bf16_kernel<<<64, 256, 0, stream>>>(W2_r, w2r_bf, (size_t)N_CLASSES * D_HID);

  // ---- degrees ----
  deg_kernel<<<(N_EDGES + 255) / 256, 256, 0, stream>>>(dst, deg, N_EDGES);
  deginv_kernel<<<(N_NODES + 255) / 256, 256, 0, stream>>>(deg, N_NODES);

  // ---- layer 1: segment-sum, mean, GEMM+ReLU ----
  {
    const long long threads = (long long)N_EDGES * 32;
    scatter_f32_kernel<<<(unsigned)((threads + 255) / 256), 256, 0, stream>>>(
        src, dst, x, aggf, N_EDGES);
  }
  scale_cvt_kernel<D_FEAT><<<ZB, 256, 0, stream>>>(aggf, deg, agg_bf,
                                                   (size_t)N_NODES * D_FEAT);
  // 6250 node-tiles x 16 col-tiles; 8 waves/block -> grid.y=2
  sage_gemm_kernel<D_FEAT, D_HID, /*RELU=*/true, /*OUT_BF16=*/true>
      <<<dim3(N_NODES / 16, 2), 256, 0, stream>>>(
          agg_bf, x_bf, w1l_bf, w1r_bf, b1, h_bf);

  // ---- layer 2: segment-sum over h, mean, GEMM -> out ----
  zero_f32_kernel<<<ZB, 256, 0, stream>>>(aggf, (size_t)N_NODES * D_HID);
  {
    const long long threads = (long long)N_EDGES * 32;
    scatter_bf16_kernel<<<(unsigned)((threads + 255) / 256), 256, 0, stream>>>(
        src, dst, h_bf, aggf, N_EDGES);
  }
  scale_cvt_kernel<D_HID><<<ZB, 256, 0, stream>>>(aggf, deg, agg_bf,
                                                  (size_t)N_NODES * D_HID);
  // 6250 node-tiles x 4 col-tiles; 4 waves/block -> grid.y=1
  sage_gemm_kernel<D_HID, N_CLASSES, /*RELU=*/false, /*OUT_BF16=*/false>
      <<<dim3(N_NODES / 16, 1), 128, 0, stream>>>(
          agg_bf, h_bf, w2l_bf, w2r_bf, b2, d_out);
}
